// BlockSparseAttention_13932873908434
// MI455X (gfx1250) — matrix operand order repaired and checked by reading, not compiled, not run on hardware
//
#include <hip/hip_runtime.h>

// Block-sparse attention for MI455X (gfx1250), wave32.
// v_wmma_f32_16x16x32_f16 for QK^T and PV. K/V tiles are prefetched with the
// CDNA5 async copy engine (global_load_async_to_lds_b128 / ASYNCcnt) into
// double-buffered f32 LDS, then converted once per tile into f16 LDS layouts
// (K row-major, V transposed) so every WMMA operand is two ds_load_b128.
// B=2 H=16 S=4096 D=64 BS=64 NB=64 W=8. One 128-thread WG per (b,h,qblock).

typedef __attribute__((ext_vector_type(16))) _Float16 v16h;
typedef __attribute__((ext_vector_type(8)))  _Float16 v8h;
typedef __attribute__((ext_vector_type(8)))  float    v8f;

#define B_   2
#define H_   16
#define S_   4096
#define D_   64
#define BS_  64
#define NB_  64
#define W_   8
#define LSTR 72   // f16 LDS row stride (halfs): 144B rows, 16B-aligned
#define VSTR 68   // f32 LDS row stride (floats): 272B rows, 16B-aligned, bank-spread

// ---- CDNA5 async global->LDS copy (ASYNCcnt) ----------------------------
__device__ inline void wait_async0() {
  asm volatile("s_wait_asynccnt 0x0" ::: "memory");
}

// Copy a 64x64 f32 tile (global row stride 64 floats) into LDS with row
// stride VSTR floats. 128 threads x 8 chunks x 16B = 16KB. GVS addressing:
// mem = SGPR64 + VGPR32. LDS dest address is taken from the vdst VGPR.
__device__ inline void async_tile_load(const float* __restrict__ gbase,
                                       uint32_t lds_base, int tid) {
#pragma unroll
  for (int c = 0; c < 8; c++) {
    const int id   = c * 128 + tid;   // 16B-chunk id, 1024 chunks per tile
    const int row  = id >> 4;         // 16 chunks per 64-float row
    const int col4 = id & 15;
    const uint32_t goff = (uint32_t)(row * (D_ * 4) + col4 * 16);
    const uint32_t loff = lds_base + (uint32_t)(row * (VSTR * 4) + col4 * 16);
    asm volatile("global_load_async_to_lds_b128 %0, %1, %2"
                 :: "v"(loff), "v"(goff), "s"(gbase)
                 : "memory");
  }
}

// ---- WMMA fragment builders (contiguous k order, shared by A and B) --------------
// A (16x32 f16): lane L (row m=L%16): halves 0..15 = K=16*(L/16)+i
//                                     (contiguous, matches B)
__device__ inline v16h load_A(const _Float16* row_base, int khalf) {
  v8h lo = *(const v8h*)(row_base + 16 * khalf);
  v8h hi = *(const v8h*)(row_base + 16 * khalf + 8);
  v16h a;
#pragma unroll
  for (int i = 0; i < 8; i++) { a[i] = lo[i]; a[8 + i] = hi[i]; }
  return a;
}

// B (32x16 f16): lane L (col n=L%16): halves 0..15 = K=16*(L/16)+0..15,
// contiguous in LDS -> two 16B loads.
__device__ inline v16h load_B_contig(const _Float16* p) {
  v8h lo = *(const v8h*)p;
  v8h hi = *(const v8h*)(p + 8);
  v16h b;
#pragma unroll
  for (int i = 0; i < 8; i++) { b[i] = lo[i]; b[8 + i] = hi[i]; }
  return b;
}

__global__ __launch_bounds__(128) void bsattn_gfx1250_kernel(
    const float* __restrict__ q, const float* __restrict__ k,
    const float* __restrict__ v, const int* __restrict__ bidx,
    float* __restrict__ out) {
  __shared__ __align__(16) float    Kbuf[2 * 64 * VSTR];  // async landing, f32 K
  __shared__ __align__(16) float    Vbuf[2 * 64 * VSTR];  // async landing, f32 V
  __shared__ __align__(16) _Float16 Kh[64 * LSTR];        // f16 K tile [kv][d]
  __shared__ __align__(16) _Float16 VhT[64 * LSTR];       // f16 V tile transposed [d][kv]
  __shared__ __align__(16) _Float16 Qsh[64 * LSTR];       // f16 Q tile
  __shared__ __align__(16) _Float16 Psh[4 * 16 * LSTR];   // per-wave f16 P tile

  const int tid    = threadIdx.x;
  const int lane   = tid & 31;
  const int wave   = tid >> 5;       // 4 waves, each owns 16 query rows
  const int lane16 = lane & 15;
  const int khalf  = lane >> 4;

  const int    nblk = blockIdx.x % NB_;
  const int    bh   = blockIdx.x / NB_;          // b*H + h
  const size_t base = (size_t)bh * S_ * D_;
  const float* qg = q + base + (size_t)nblk * BS_ * D_;
  float*       og = out + base + (size_t)nblk * BS_ * D_;

  const uint32_t kls[2] = { (uint32_t)(uintptr_t)(Kbuf),
                            (uint32_t)(uintptr_t)(Kbuf + 64 * VSTR) };
  const uint32_t vls[2] = { (uint32_t)(uintptr_t)(Vbuf),
                            (uint32_t)(uintptr_t)(Vbuf + 64 * VSTR) };

  const int row = tid >> 1, sub = tid & 1;  // staging/convert mapping: 2 thr/row

  // ---- stage Q (once, f16) + kick off async prefetch of KV block 0 ----
  {
    const float4* gp = (const float4*)(qg + row * D_ + sub * 32);
    alignas(16) _Float16 tmp[32];
#pragma unroll
    for (int i = 0; i < 8; i++) {
      float4 f = gp[i];
      tmp[4 * i + 0] = (_Float16)f.x; tmp[4 * i + 1] = (_Float16)f.y;
      tmp[4 * i + 2] = (_Float16)f.z; tmp[4 * i + 3] = (_Float16)f.w;
    }
    v8h* sp = (v8h*)(Qsh + row * LSTR + sub * 32);
#pragma unroll
    for (int i = 0; i < 4; i++) sp[i] = *(const v8h*)(tmp + 8 * i);
  }
  {
    const int kb0 = bidx[nblk * W_];
    async_tile_load(k + base + (size_t)kb0 * BS_ * D_, kls[0], tid);
    async_tile_load(v + base + (size_t)kb0 * BS_ * D_, vls[0], tid);
  }
  __syncthreads();  // Qsh visible to all waves

  // Q A-fragments for this wave's 16 rows, K-chunks d=[0,32) and [32,64)
  v16h qa[2];
  {
    const _Float16* rp = Qsh + (16 * wave + lane16) * LSTR;
    qa[0] = load_A(rp + 0, khalf);
    qa[1] = load_A(rp + 32, khalf);
  }

  // Running softmax state. C-layout: VGPR r holds row m = r + 8*(lane/16), col n = lane%16.
  float m_run[8], l_run[8];
  v8f acc[4];
#pragma unroll
  for (int r = 0; r < 8; r++) { m_run[r] = -1e30f; l_run[r] = 0.0f; }
#pragma unroll
  for (int t = 0; t < 4; t++) { v8f z = {}; acc[t] = z; }

  const float scale = 0.125f;  // 1/sqrt(64)
  _Float16* pw = Psh + wave * 16 * LSTR;

  for (int j = 0; j < W_; j++) {
    const int cur = j & 1;

    // Own async copies for buffer `cur` landed, then WG-wide sync: also
    // guarantees all waves finished last iteration's reads of Kh/VhT and of
    // the f32 buffer we are about to overwrite.
    wait_async0();
    __syncthreads();

    if (j + 1 < W_) {  // prefetch next KV block into the other f32 buffer
      const int kbn = bidx[nblk * W_ + j + 1];
      async_tile_load(k + base + (size_t)kbn * BS_ * D_, kls[cur ^ 1], tid);
      async_tile_load(v + base + (size_t)kbn * BS_ * D_, vls[cur ^ 1], tid);
    }

    const float* Kcur = Kbuf + cur * 64 * VSTR;
    const float* Vcur = Vbuf + cur * 64 * VSTR;

    // ---- one-time convert pass (overlapped with async prefetch) ----
    {   // K: f32 -> f16, same row-major layout
      const float4* p4 = (const float4*)(Kcur + row * VSTR + sub * 32);
      alignas(16) _Float16 tmp[32];
#pragma unroll
      for (int i = 0; i < 8; i++) {
        float4 f = p4[i];
        tmp[4 * i + 0] = (_Float16)f.x; tmp[4 * i + 1] = (_Float16)f.y;
        tmp[4 * i + 2] = (_Float16)f.z; tmp[4 * i + 3] = (_Float16)f.w;
      }
      v8h* sp = (v8h*)(Kh + row * LSTR + sub * 32);
#pragma unroll
      for (int i = 0; i < 4; i++) sp[i] = *(const v8h*)(tmp + 8 * i);
    }
    {   // V: f32 -> f16 with transpose: VhT[d][kv] = V[kv][d]
      const float4* p4 = (const float4*)(Vcur + row * VSTR + sub * 32);
#pragma unroll
      for (int i = 0; i < 8; i++) {
        float4 f = p4[i];
        const int d = sub * 32 + 4 * i;
        VhT[(d + 0) * LSTR + row] = (_Float16)f.x;
        VhT[(d + 1) * LSTR + row] = (_Float16)f.y;
        VhT[(d + 2) * LSTR + row] = (_Float16)f.z;
        VhT[(d + 3) * LSTR + row] = (_Float16)f.w;
      }
    }
    __syncthreads();

    // ---- S = (Q K^T) * scale : 4 tiles of 16x16, K-dim = d (2 chunks of 32) ----
    v8f s[4];
#pragma unroll
    for (int t = 0; t < 4; t++) {
      v8f c = {};
#pragma unroll
      for (int ch = 0; ch < 2; ch++) {
        const _Float16* bp = Kh + (16 * t + lane16) * LSTR + 32 * ch + 16 * khalf;
        v16h b = load_B_contig(bp);
        c = __builtin_amdgcn_wmma_f32_16x16x32_f16(false, qa[ch], false, b,
                                                   (short)0, c, false, false);
      }
#pragma unroll
      for (int r = 0; r < 8; r++) c[r] *= scale;
      s[t] = c;
    }

    // ---- online softmax update (xor-shuffles stay within 16-lane halves) ----
    float alpha[8];
#pragma unroll
    for (int r = 0; r < 8; r++) {
      float mx = fmaxf(fmaxf(s[0][r], s[1][r]), fmaxf(s[2][r], s[3][r]));
#pragma unroll
      for (int off = 1; off < 16; off <<= 1)
        mx = fmaxf(mx, __shfl_xor(mx, off, 32));
      const float mnew = fmaxf(m_run[r], mx);
      alpha[r] = __expf(m_run[r] - mnew);
      m_run[r] = mnew;

      float rs = 0.0f;
#pragma unroll
      for (int t = 0; t < 4; t++) {
        float p = __expf(s[t][r] - mnew);
        s[t][r] = p;
        rs += p;
      }
#pragma unroll
      for (int off = 1; off < 16; off <<= 1)
        rs += __shfl_xor(rs, off, 32);
      l_run[r] = l_run[r] * alpha[r] + rs;
    }

    // ---- write P (f16) to wave-private LDS; rescale O accumulators ----
#pragma unroll
    for (int t = 0; t < 4; t++) {
#pragma unroll
      for (int r = 0; r < 8; r++) {
        pw[(r + 8 * khalf) * LSTR + 16 * t + lane16] = (_Float16)s[t][r];
        acc[t][r] *= alpha[r];
      }
    }
    // LDS ops of one wave complete in order -> safe to re-gather without barrier.

    // ---- O += P * V : A = P (16 x kv), B = V^T rows (contiguous kv) ----
    v16h pa[2];
    {
      const _Float16* prow = pw + lane16 * LSTR;
      pa[0] = load_A(prow + 0, khalf);
      pa[1] = load_A(prow + 32, khalf);
    }
#pragma unroll
    for (int t = 0; t < 4; t++) {
#pragma unroll
      for (int ch = 0; ch < 2; ch++) {
        const _Float16* vp = VhT + (16 * t + lane16) * LSTR + 32 * ch + 16 * khalf;
        v16h bv = load_B_contig(vp);
        acc[t] = __builtin_amdgcn_wmma_f32_16x16x32_f16(false, pa[ch], false, bv,
                                                        (short)0, acc[t], false, false);
      }
    }
  }

  // ---- epilogue: O / l ----
#pragma unroll
  for (int r = 0; r < 8; r++) {
    const float inv_l = 1.0f / l_run[r];
    const int m = 16 * wave + r + 8 * khalf;
#pragma unroll
    for (int t = 0; t < 4; t++)
      og[m * D_ + 16 * t + lane16] = acc[t][r] * inv_l;
  }
}

extern "C" void kernel_launch(void* const* d_in, const int* in_sizes, int n_in,
                              void* d_out, int out_size, void* d_ws, size_t ws_size,
                              hipStream_t stream) {
  (void)in_sizes; (void)n_in; (void)d_ws; (void)ws_size; (void)out_size;
  const float* q    = (const float*)d_in[0];
  const float* k    = (const float*)d_in[1];
  const float* v    = (const float*)d_in[2];
  const int*   bidx = (const int*)d_in[3];
  float*       out  = (float*)d_out;

  dim3 grid(B_ * H_ * NB_);   // 2048 workgroups: one per (b,h,qblock)
  dim3 block(128);            // 4 wave32
  bsattn_gfx1250_kernel<<<grid, block, 0, stream>>>(q, k, v, bidx, out);
}
